// DescentPredictor_26319559590183
// MI455X (gfx1250) — compile-verified, matrix operand order checked by reading, not run on hardware
//
#include <hip/hip_runtime.h>
#include <cstdint>

typedef unsigned short u16;
typedef __attribute__((ext_vector_type(16))) __bf16 v16bf;
typedef __attribute__((ext_vector_type(8)))  float  v8f;

#define NROWS 16384
#define LRATE 0.1f
#define NITER 50
#define LNEPS 1e-5f

// ---- workspace layout (units: u16 elements) ----
// c0 [16384][256] bf16, then fragment-swizzled bf16 weights.
#define OFF_C0    0u
#define OFF_W0XT  4194304u                 // 512x256  fwd (B[k][n]=W0[n][k])
#define OFF_W0YT  (OFF_W0XT + 131072u)     // 64x256   fwd (B[k][n]=W0[n][512+k])
#define OFF_W1T   (OFF_W0YT + 16384u)      // 256x256  fwd
#define OFF_W2T   (OFF_W1T  + 65536u)      // 256x256  fwd
#define OFF_W1B   (OFF_W2T  + 65536u)      // 256x256  bwd (B[k][n]=W1[k][n])
#define OFF_W2B   (OFF_W1B  + 65536u)      // 256x256  bwd
#define OFF_W0YB  (OFF_W2B  + 65536u)      // 256x64   bwd (B[k][n]=W0[k][512+n])
#define SWZ_TOTAL (131072u + 16384u + 4u*65536u + 16384u)

__device__ __forceinline__ u16 f2bf(float f) {
    union { float f; uint32_t u; } v; v.f = f;
    uint32_t r = v.u + 0x7FFFu + ((v.u >> 16) & 1u);   // round-to-nearest-even
    return (u16)(r >> 16);
}
__device__ __forceinline__ float bf2f(u16 h) {
    union { uint32_t u; float f; } v; v.u = ((uint32_t)h) << 16;
    return v.f;
}

// Fragment convention (shared by swizzler + consumers):
// fragment f = nt*NKT + kt holds B[kt*32 .. +31][nt*16 .. +15];
// lane L stores 16 consecutive-k bf16: k = kt*32 + (L>>4)*16 + e, n = nt*16 + (L&15).
__device__ __forceinline__ void frag_kn(unsigned rel, int nkt, int& k, int& n) {
    unsigned f = rel >> 9, r = rel & 511u, lane = r >> 4, e = r & 15u;
    unsigned kt = f % (unsigned)nkt, nt = f / (unsigned)nkt;
    k = (int)(kt * 32u + (lane >> 4) * 16u + e);
    n = (int)(nt * 16u + (lane & 15u));
}

// D = A[16xK] * B[KxN-tile] + acc, bf16 inputs, f32 accum. A from row-major u16
// buffer (LDS), B from fragment-swizzled buffer (LDS or global).
template<int NKT, int NNT>
__device__ __forceinline__ void wave_mm(const u16* aBuf, int aStride,
                                        const u16* bBase, int lane, v8f* acc) {
    const int m = lane & 15, hx = lane >> 4;
#pragma unroll
    for (int nt = 0; nt < NNT; ++nt) {
        v8f c = acc[nt];
#pragma unroll
        for (int kt = 0; kt < NKT; ++kt) {
            v16bf a = *(const v16bf*)&aBuf[m * aStride + kt * 32 + hx * 16];
            v16bf b = *(const v16bf*)&bBase[((nt * NKT + kt) * 32 + lane) * 16];
            c = __builtin_amdgcn_wmma_f32_16x16x32_bf16(false, a, false, b,
                                                        (short)0, c, false, false);
        }
        acc[nt] = c;
    }
}

// Cooperative 64-thread weight staging into the LDS slot (b128 copies).
__device__ __forceinline__ void stageW(u16* dst, const u16* src, int nU16, int tid) {
    __syncthreads();
    const uint4* s = (const uint4*)src;
    uint4* d = (uint4*)dst;
    int n16 = nU16 >> 3;
    for (int i = tid; i < n16; i += 64) d[i] = s[i];
    __syncthreads();
}

// LayerNorm + SiLU forward on 16x256 D-tiles; writes zhat (bf16) + h (bf16, A layout)
// and rstd. Row stats via shfl_xor butterfly over the 16-lane halves.
__device__ __forceinline__ void ln_fwd(v8f* acc, int lane, const float* __restrict__ g,
                                       const float* __restrict__ be,
                                       u16* zhat, u16* hbuf, float* rstd) {
    const int m15 = lane & 15, hx = lane >> 4;
    float sum[8], sq[8];
#pragma unroll
    for (int r = 0; r < 8; ++r) { sum[r] = 0.f; sq[r] = 0.f; }
#pragma unroll
    for (int nt = 0; nt < 16; ++nt)
#pragma unroll
        for (int r = 0; r < 8; ++r) { float v = acc[nt][r]; sum[r] += v; sq[r] += v * v; }
#pragma unroll
    for (int d = 1; d < 16; d <<= 1)
#pragma unroll
        for (int r = 0; r < 8; ++r) {
            sum[r] += __shfl_xor(sum[r], d, 32);
            sq[r]  += __shfl_xor(sq[r],  d, 32);
        }
    float mu[8], rs[8];
#pragma unroll
    for (int r = 0; r < 8; ++r) {
        mu[r] = sum[r] * (1.f / 256.f);
        float var = sq[r] * (1.f / 256.f) - mu[r] * mu[r];
        rs[r] = rsqrtf(var + LNEPS);
    }
    if (m15 == 0) {
#pragma unroll
        for (int r = 0; r < 8; ++r) rstd[hx * 8 + r] = rs[r];
    }
#pragma unroll
    for (int nt = 0; nt < 16; ++nt) {
        int n = nt * 16 + m15; float gv = g[n], bv = be[n];
#pragma unroll
        for (int r = 0; r < 8; ++r) {
            float zh = (acc[nt][r] - mu[r]) * rs[r];
            float s  = gv * zh + bv;
            float sg = 1.f / (1.f + __expf(-s));
            int mm = hx * 8 + r;
            zhat[mm * 256 + n] = f2bf(zh);
            hbuf[mm * 256 + n] = f2bf(s * sg);
        }
    }
}

// Backward through SiLU + LayerNorm: acc holds dh in, dz out.
__device__ __forceinline__ void ln_bwd(v8f* acc, int lane, const u16* zhat,
                                       const float* __restrict__ g,
                                       const float* __restrict__ be,
                                       const float* rstd) {
    const int m15 = lane & 15, hx = lane >> 4;
    float s1[8], s2[8];
#pragma unroll
    for (int r = 0; r < 8; ++r) { s1[r] = 0.f; s2[r] = 0.f; }
#pragma unroll
    for (int nt = 0; nt < 16; ++nt) {
        int n = nt * 16 + m15; float gv = g[n], bv = be[n];
#pragma unroll
        for (int r = 0; r < 8; ++r) {
            int mm = hx * 8 + r;
            float zh = bf2f(zhat[mm * 256 + n]);
            float s  = gv * zh + bv;
            float sg = 1.f / (1.f + __expf(-s));
            float dsil = sg * (1.f + s * (1.f - sg));
            float dzh = acc[nt][r] * dsil * gv;
            acc[nt][r] = dzh;
            s1[r] += dzh; s2[r] += dzh * zh;
        }
    }
#pragma unroll
    for (int d = 1; d < 16; d <<= 1)
#pragma unroll
        for (int r = 0; r < 8; ++r) {
            s1[r] += __shfl_xor(s1[r], d, 32);
            s2[r] += __shfl_xor(s2[r], d, 32);
        }
#pragma unroll
    for (int r = 0; r < 8; ++r) { s1[r] *= (1.f / 256.f); s2[r] *= (1.f / 256.f); }
#pragma unroll
    for (int nt = 0; nt < 16; ++nt) {
        int n = nt * 16 + m15;
#pragma unroll
        for (int r = 0; r < 8; ++r) {
            int mm = hx * 8 + r;
            float zh = bf2f(zhat[mm * 256 + n]);
            acc[nt][r] = rstd[mm] * (acc[nt][r] - s1[r] - zh * s2[r]);
        }
    }
}

__device__ __forceinline__ void store_dz(const v8f* acc, int lane, u16* hbuf) {
    const int m15 = lane & 15, hx = lane >> 4;
#pragma unroll
    for (int nt = 0; nt < 16; ++nt) {
        int n = nt * 16 + m15;
#pragma unroll
        for (int r = 0; r < 8; ++r) hbuf[(hx * 8 + r) * 256 + n] = f2bf(acc[nt][r]);
    }
}

// ---------------- Kernel A: bf16 conversion + fragment swizzle ----------------
__global__ void ebm_swz_kernel(const float* __restrict__ W0, const float* __restrict__ W1,
                               const float* __restrict__ W2, u16* __restrict__ ws) {
    unsigned idx = blockIdx.x * 256u + threadIdx.x;
    if (idx >= SWZ_TOTAL) return;
    unsigned rel = idx; int k, n; float v; unsigned dst;
    if (rel < 131072u)      { frag_kn(rel, 16, k, n); v = W0[n * 576 + k];        dst = OFF_W0XT + rel; }
    else { rel -= 131072u;
    if (rel < 16384u)       { frag_kn(rel, 2,  k, n); v = W0[n * 576 + 512 + k];  dst = OFF_W0YT + rel; }
    else { rel -= 16384u;
    if (rel < 65536u)       { frag_kn(rel, 8,  k, n); v = W1[n * 256 + k];        dst = OFF_W1T + rel; }
    else { rel -= 65536u;
    if (rel < 65536u)       { frag_kn(rel, 8,  k, n); v = W2[n * 256 + k];        dst = OFF_W2T + rel; }
    else { rel -= 65536u;
    if (rel < 65536u)       { frag_kn(rel, 8,  k, n); v = W1[k * 256 + n];        dst = OFF_W1B + rel; }
    else { rel -= 65536u;
    if (rel < 65536u)       { frag_kn(rel, 8,  k, n); v = W2[k * 256 + n];        dst = OFF_W2B + rel; }
    else { rel -= 65536u;     frag_kn(rel, 8,  k, n); v = W0[k * 576 + 512 + n];  dst = OFF_W0YB + rel; }
    }}}}}
    ws[dst] = f2bf(v);
}

// ---------------- Kernel B: c0 = x @ W0x^T + b0 (bf16 WMMA) ----------------
__global__ __launch_bounds__(64) void ebm_c0_kernel(const float* __restrict__ x,
                                                    const float* __restrict__ b0,
                                                    const u16* __restrict__ ws,
                                                    u16* __restrict__ c0out) {
    __shared__ __align__(32) u16 sX[2][16 * 512];
    int tid = threadIdx.x, lane = tid & 31, w = tid >> 5;
    int rowBase = blockIdx.x * 32 + w * 16;
    for (int i = lane; i < 16 * 512; i += 32)
        sX[w][i] = f2bf(x[(rowBase + (i >> 9)) * 512 + (i & 511)]);
    const int m15 = lane & 15, hx = lane >> 4;
    v8f acc[16];
#pragma unroll
    for (int nt = 0; nt < 16; ++nt) {
        float bv = b0[nt * 16 + m15];
#pragma unroll
        for (int r = 0; r < 8; ++r) acc[nt][r] = bv;
    }
    wave_mm<16, 16>(sX[w], 512, ws + OFF_W0XT, lane, acc);   // B frags from L2
#pragma unroll
    for (int nt = 0; nt < 16; ++nt) {
        int n = nt * 16 + m15;
#pragma unroll
        for (int r = 0; r < 8; ++r)
            c0out[(unsigned)(rowBase + hx * 8 + r) * 256u + n] = f2bf(acc[nt][r]);
    }
}

// ---------------- Kernel C: persistent per-tile gradient descent ----------------
__global__ __launch_bounds__(64) void ebm_descent_kernel(
    const float* __restrict__ y0,
    const float* __restrict__ g0, const float* __restrict__ be0,
    const float* __restrict__ b1, const float* __restrict__ g1, const float* __restrict__ be1,
    const float* __restrict__ b2, const float* __restrict__ g2, const float* __restrict__ be2,
    const float* __restrict__ wout,
    const u16* __restrict__ ws, float* __restrict__ yout) {

    __shared__ __align__(32) u16  sW[32768];          // 64KB weight slot
    __shared__ __align__(32) u16  sZhat[2][3][4096];  // per-layer zhat, bf16
    __shared__ __align__(32) u16  sH[2][4096];        // A-operand staging, bf16
    __shared__ __align__(32) u16  sC0[2][4096];       // c0 tile, bf16
    __shared__ float sY[2][1024];                     // live y tile, f32
    __shared__ float sRstd[2][3][16];

    int tid = threadIdx.x, lane = tid & 31, w = tid >> 5;
    int rowBase = blockIdx.x * 32 + w * 16;
    const int m15 = lane & 15, hx = lane >> 4;
    const float invB = 1.0f / (float)NROWS;

    for (int i = lane; i < 1024; i += 32)
        sY[w][i] = y0[(rowBase + (i >> 6)) * 64 + (i & 63)];
    for (int i = lane; i < 4096; i += 32)
        sC0[w][i] = ws[OFF_C0 + (unsigned)(rowBase + (i >> 8)) * 256u + (i & 255)];

    __builtin_prefetch(ws + OFF_W0YT, 0, 0);

#pragma unroll 1
    for (int t = 0; t < NITER; ++t) {
        v8f acc[16];

        // ---- z0 = c0 + y @ W0y^T ----
        stageW(sW, ws + OFF_W0YT, 16384, tid);
        for (int i = lane; i < 1024; i += 32) sH[w][i] = f2bf(sY[w][i]);  // y as A [16][64]
#pragma unroll
        for (int nt = 0; nt < 16; ++nt) {
            int n = nt * 16 + m15;
#pragma unroll
            for (int r = 0; r < 8; ++r)
                acc[nt][r] = bf2f(sC0[w][(hx * 8 + r) * 256 + n]);
        }
        wave_mm<2, 16>(sH[w], 64, sW, lane, acc);
        ln_fwd(acc, lane, g0, be0, sZhat[w][0], sH[w], sRstd[w][0]);

        // ---- z1 = h1 @ W1^T + b1 (two 64KB half-stages) ----
        __builtin_prefetch(ws + OFF_W1T, 0, 0);
#pragma unroll
        for (int nt = 0; nt < 16; ++nt) {
            float bv = b1[nt * 16 + m15];
#pragma unroll
            for (int r = 0; r < 8; ++r) acc[nt][r] = bv;
        }
        stageW(sW, ws + OFF_W1T, 32768, tid);
        wave_mm<8, 8>(sH[w], 256, sW, lane, acc);
        stageW(sW, ws + OFF_W1T + 32768, 32768, tid);
        wave_mm<8, 8>(sH[w], 256, sW, lane, acc + 8);
        ln_fwd(acc, lane, g1, be1, sZhat[w][1], sH[w], sRstd[w][1]);

        // ---- z2 = h2 @ W2^T + b2 ----
#pragma unroll
        for (int nt = 0; nt < 16; ++nt) {
            float bv = b2[nt * 16 + m15];
#pragma unroll
            for (int r = 0; r < 8; ++r) acc[nt][r] = bv;
        }
        stageW(sW, ws + OFF_W2T, 32768, tid);
        wave_mm<8, 8>(sH[w], 256, sW, lane, acc);
        stageW(sW, ws + OFF_W2T + 32768, 32768, tid);
        wave_mm<8, 8>(sH[w], 256, sW, lane, acc + 8);
        ln_fwd(acc, lane, g2, be2, sZhat[w][2], sH[w], sRstd[w][2]);

        // ---- backward: dh3 = wout/B (linear head, value of E not needed) ----
#pragma unroll
        for (int nt = 0; nt < 16; ++nt) {
            float wv = wout[nt * 16 + m15] * invB;
#pragma unroll
            for (int r = 0; r < 8; ++r) acc[nt][r] = wv;
        }
        ln_bwd(acc, lane, sZhat[w][2], g2, be2, sRstd[w][2]);  // -> dz3
        store_dz(acc, lane, sH[w]);

        // ---- dh2 = dz3 @ W2 ----
#pragma unroll
        for (int nt = 0; nt < 16; ++nt)
#pragma unroll
            for (int r = 0; r < 8; ++r) acc[nt][r] = 0.f;
        stageW(sW, ws + OFF_W2B, 32768, tid);
        wave_mm<8, 8>(sH[w], 256, sW, lane, acc);
        stageW(sW, ws + OFF_W2B + 32768, 32768, tid);
        wave_mm<8, 8>(sH[w], 256, sW, lane, acc + 8);
        ln_bwd(acc, lane, sZhat[w][1], g1, be1, sRstd[w][1]);  // -> dz2
        store_dz(acc, lane, sH[w]);

        // ---- dh1 = dz2 @ W1 ----
#pragma unroll
        for (int nt = 0; nt < 16; ++nt)
#pragma unroll
            for (int r = 0; r < 8; ++r) acc[nt][r] = 0.f;
        stageW(sW, ws + OFF_W1B, 32768, tid);
        wave_mm<8, 8>(sH[w], 256, sW, lane, acc);
        stageW(sW, ws + OFF_W1B + 32768, 32768, tid);
        wave_mm<8, 8>(sH[w], 256, sW, lane, acc + 8);
        ln_bwd(acc, lane, sZhat[w][0], g0, be0, sRstd[w][0]);  // -> dz1
        store_dz(acc, lane, sH[w]);

        // ---- dy = dz1 @ W0y ; y -= LR * dy ----
        stageW(sW, ws + OFF_W0YB, 16384, tid);
        v8f dyacc[4];
#pragma unroll
        for (int nt = 0; nt < 4; ++nt)
#pragma unroll
            for (int r = 0; r < 8; ++r) dyacc[nt][r] = 0.f;
        wave_mm<8, 4>(sH[w], 256, sW, lane, dyacc);
#pragma unroll
        for (int nt = 0; nt < 4; ++nt) {
            int n = nt * 16 + m15;
#pragma unroll
            for (int r = 0; r < 8; ++r)
                sY[w][(hx * 8 + r) * 64 + n] -= LRATE * dyacc[nt][r];
        }
    }

    for (int i = lane; i < 1024; i += 32)
        yout[(rowBase + (i >> 6)) * 64 + (i & 63)] = sY[w][i];
}

extern "C" void kernel_launch(void* const* d_in, const int* in_sizes, int n_in,
                              void* d_out, int out_size, void* d_ws, size_t ws_size,
                              hipStream_t stream) {
    (void)in_sizes; (void)n_in; (void)out_size; (void)ws_size;
    const float* x    = (const float*)d_in[0];
    const float* y0   = (const float*)d_in[1];
    const float* W0   = (const float*)d_in[2];
    const float* b0   = (const float*)d_in[3];
    const float* g0   = (const float*)d_in[4];
    const float* be0  = (const float*)d_in[5];
    const float* W1   = (const float*)d_in[6];
    const float* b1   = (const float*)d_in[7];
    const float* g1   = (const float*)d_in[8];
    const float* be1  = (const float*)d_in[9];
    const float* W2   = (const float*)d_in[10];
    const float* b2   = (const float*)d_in[11];
    const float* g2   = (const float*)d_in[12];
    const float* be2  = (const float*)d_in[13];
    const float* wout = (const float*)d_in[14];
    u16*   ws   = (u16*)d_ws;
    float* yout = (float*)d_out;

    ebm_swz_kernel<<<(SWZ_TOTAL + 255u) / 256u, 256, 0, stream>>>(W0, W1, W2, ws);
    ebm_c0_kernel<<<NROWS / 32, 64, 0, stream>>>(x, b0, ws, ws + OFF_C0);
    ebm_descent_kernel<<<NROWS / 32, 64, 0, stream>>>(y0, g0, be0, b1, g1, be1,
                                                      b2, g2, be2, wout, ws, yout);
}
